// DecoderLayer_14663018348828
// MI455X (gfx1250) — matrix operand order repaired and checked by reading, not compiled, not run on hardware
//
#include <hip/hip_runtime.h>
#include <hip/hip_bf16.h>

typedef unsigned short u16;
typedef unsigned int   u32;
typedef unsigned long long u64;
typedef __bf16 bf16;
typedef bf16  v16bf __attribute__((ext_vector_type(16)));
typedef float v8f   __attribute__((ext_vector_type(8)));
typedef u32   v8u   __attribute__((ext_vector_type(8)));

#define DM    1024
#define DF    4096
#define NBATCH 4
#define SLEN  4096
#define MROWS (NBATCH * SLEN)   // 16384
#define EPSF  1e-6f

// ---------------------------------------------------------------- helpers
__device__ __forceinline__ u16 f2bf(float f) {
  u32 u = __builtin_bit_cast(u32, f);
  u32 r = (u + 0x7FFFu + ((u >> 16) & 1u)) >> 16;   // round-to-nearest-even
  return (u16)r;
}

__device__ __forceinline__ void async_wait0() {
  asm volatile("s_wait_asynccnt 0" ::: "memory");
}

// one 16-byte chunk, global memory -> LDS, tracked by ASYNCcnt
__device__ __forceinline__ void async_b128(u32 lds_addr, const void* gptr) {
  asm volatile("global_load_async_to_lds_b128 %0, %1, off"
               :: "v"(lds_addr), "v"((u64)gptr) : "memory");
}

// ---------------------------------------------------------------- elementwise
__global__ void add_cvt_kernel(const float* __restrict__ a, const float* __restrict__ b,
                               float* __restrict__ of, u16* __restrict__ ob, size_t n) {
  size_t i = (size_t)blockIdx.x * blockDim.x + threadIdx.x;
  size_t stride = (size_t)gridDim.x * blockDim.x;
  for (; i < n; i += stride) {
    float v = a[i] + b[i];
    of[i] = v;
    ob[i] = f2bf(v);
  }
}

// transpose + convert: in f32 [K][N]  ->  out bf16 [N][K]   (64x64 tiles)
__global__ __launch_bounds__(256) void cvt_transpose_kernel(
    const float* __restrict__ in, u16* __restrict__ out, int K, int N) {
  __shared__ u16 tile[64][65];
  const int t  = threadIdx.x;
  const int bn = blockIdx.x;   // tile along N
  const int bk = blockIdx.y;   // tile along K
#pragma unroll
  for (int j = 0; j < 16; ++j) {
    const int idx = t + j * 256;
    const int r = idx >> 6, c = idx & 63;                 // r: k, c: n
    tile[c][r] = f2bf(in[(size_t)(bk * 64 + r) * N + (size_t)(bn * 64 + c)]);
  }
  __syncthreads();
#pragma unroll
  for (int j = 0; j < 16; ++j) {
    const int idx = t + j * 256;
    const int r = idx >> 6, c = idx & 63;                 // r: n, c: k
    out[(size_t)(bn * 64 + r) * K + (size_t)(bk * 64 + c)] = tile[r][c];
  }
}

// ---------------------------------------------------------------- WMMA GEMM
// C[M,N] = act(A[M,K] * B[K,N] + bias); A bf16 row-major [M,K], B given TRANSPOSED
// as bf16 [N,K].  Block: 256 threads / 8 waves; block tile 128x256, wave tile
// 64x64 = 4x4 x (v_wmma_f32_16x16x32_bf16).  Panels staged in double-buffered
// LDS via GLOBAL_LOAD_ASYNC_TO_LDS_B128 (ASYNCcnt).
#define BKM 128
#define BKN 256
#define BKK 32
#define LDP 40   // halfwords/row: 32 data + 8 pad -> 80B rows, 16B-aligned chunks,
                 // conflict-free 16-lane fragment reads ((n*20) mod 64 distinct)

__device__ __forceinline__ void issue_async_panels(
    const u16* __restrict__ A, const u16* __restrict__ BT,
    size_t aRow0, int n0, int k0, int K, u32 ldsA, u32 ldsB, int t) {
#pragma unroll
  for (int i = 0; i < 2; ++i) {            // A panel: 128 rows x 32 hw = 512 chunks
    const int idx = t + i * 256;
    const int row = idx >> 2, ch = (idx & 3) * 8;
    async_b128(ldsA + (u32)(row * LDP + ch) * 2u,
               A + (aRow0 + (size_t)row) * (size_t)K + (size_t)(k0 + ch));
  }
#pragma unroll
  for (int i = 0; i < 4; ++i) {            // B panel: 256 rows x 32 hw = 1024 chunks
    const int idx = t + i * 256;
    const int row = idx >> 2, ch = (idx & 3) * 8;
    async_b128(ldsB + (u32)(row * LDP + ch) * 2u,
               BT + (size_t)(n0 + row) * (size_t)K + (size_t)(k0 + ch));
  }
}

__global__ __launch_bounds__(256, 1) void gemm_bf16_kernel(
    const u16* __restrict__ A, const u16* __restrict__ BT,
    const float* __restrict__ bias, float* __restrict__ Cf, u16* __restrict__ Cb,
    int Mr, int N, int K, int act) {
  __shared__ u16 sA[2][BKM][LDP];          // 20 KB
  __shared__ u16 sB[2][BKN][LDP];          // 40 KB

  const int t    = threadIdx.x;
  const int lane = t & 31;
  const int wid  = t >> 5;
  const int wm   = wid >> 2;        // 0..1
  const int wn   = wid & 3;         // 0..3
  const int l16  = lane & 15;
  const int lh   = lane >> 4;       // 0/1

  const size_t aRow0 = (size_t)blockIdx.y * BKM;
  const int    n0    = blockIdx.x * BKN;

  const u32 ldsA0 = (u32)(u64)&sA[0][0][0];
  const u32 ldsB0 = (u32)(u64)&sB[0][0][0];
  const u32 stA   = (u32)(BKM * LDP * 2);
  const u32 stB   = (u32)(BKN * LDP * 2);

  v8f acc[4][4];
#pragma unroll
  for (int mi = 0; mi < 4; ++mi)
#pragma unroll
    for (int ni = 0; ni < 4; ++ni)
#pragma unroll
      for (int r = 0; r < 8; ++r) acc[mi][ni][r] = 0.0f;

  issue_async_panels(A, BT, aRow0, n0, 0, K, ldsA0, ldsB0, t);
  async_wait0();
  __syncthreads();

  const int ksteps = K / BKK;
  for (int kt = 0; kt < ksteps; ++kt) {
    const int cur = kt & 1;
    if (kt + 1 < ksteps)   // async-copy next panel into the other buffer
      issue_async_panels(A, BT, aRow0, n0, (kt + 1) * BKK, K,
                         ldsA0 + (u32)(cur ^ 1) * stA, ldsB0 + (u32)(cur ^ 1) * stB, t);
    if (kt + 2 < ksteps) { // L2 prefetch hint two panels ahead
      __builtin_prefetch(A + (aRow0 + (t >> 3)) * (size_t)K + (size_t)(kt + 2) * BKK, 0, 1);
      __builtin_prefetch(BT + (size_t)(n0 + (t >> 5)) * (size_t)K + (size_t)(kt + 2) * BKK, 0, 1);
    }

    // B fragments (contiguous k order, same as A: lanes0-15 K=0..15, lanes16-31 K=16..31)
    v16bf bfrag[4];
#pragma unroll
    for (int ni = 0; ni < 4; ++ni) {
      const int n = wn * 64 + ni * 16 + l16;
      v8u bu;
#pragma unroll
      for (int i = 0; i < 8; ++i)
        bu[i] = *(const u32*)&sB[cur][n][2 * i + 16 * lh];
      bfrag[ni] = __builtin_bit_cast(v16bf, bu);
    }
#pragma unroll
    for (int mi = 0; mi < 4; ++mi) {
      // A fragment (contiguous k = 16*lh + s, same order as B)
      const int m = wm * 64 + mi * 16 + l16;
      v8u au;
#pragma unroll
      for (int i = 0; i < 8; ++i) {
        const int kk = 2 * i + 16 * lh;
        au[i] = *(const u32*)&sA[cur][m][kk];
      }
      const v16bf afrag = __builtin_bit_cast(v16bf, au);
#pragma unroll
      for (int ni = 0; ni < 4; ++ni)
        acc[mi][ni] = __builtin_amdgcn_wmma_f32_16x16x32_bf16(
            false, afrag, false, bfrag[ni], (short)0, acc[mi][ni], false, false);
    }

    if (kt + 1 < ksteps) async_wait0();    // this wave's async copies landed
    __syncthreads();                       // all waves' copies / reads ordered
  }

  // epilogue: C/D layout = vgpr r -> row r + 8*lh, lane -> col l16
#pragma unroll
  for (int mi = 0; mi < 4; ++mi) {
#pragma unroll
    for (int ni = 0; ni < 4; ++ni) {
      const size_t row0 = aRow0 + (size_t)(wm * 64 + mi * 16);
      const int    col  = n0 + wn * 64 + ni * 16 + l16;
      const float  bv   = bias ? bias[col] : 0.0f;
#pragma unroll
      for (int r = 0; r < 8; ++r) {
        const size_t row = row0 + (size_t)(r + 8 * lh);
        float v = acc[mi][ni][r] + bv;
        if (act == 1) v = (v > 0.0f) ? (v + 1.0f) : __expf(v);     // elu(v)+1
        else if (act == 2) { v = fmaxf(v, 0.0f); v = v * v; }       // relu^2
        const size_t o = row * (size_t)N + (size_t)col;
        if (Cf) Cf[o] = v;
        if (Cb) Cb[o] = f2bf(v);
      }
    }
  }
}

// ---------------------------------------------------------------- linear-attn scan
// one thread per (batch, channel); coalesced across channels each step.
__global__ void scan_kernel(const float* q, const float* k, const float* v, float* attn) {
  const int t = blockIdx.x * blockDim.x + threadIdx.x;   // 0..4095
  const int b = t >> 10;
  const int e = t & (DM - 1);
  size_t idx = (size_t)b * SLEN * DM + (size_t)e;
  float rk = 0.0f, rkv = 0.0f;
  for (int s = 0; s < SLEN; ++s) {
    const float kv_ = k[idx] * v[idx];
    rk  += k[idx];
    rkv += kv_;
    attn[idx] = q[idx] * ((kv_ + rkv) / rk);   // attn may alias q (same index)
    idx += DM;
  }
}

// ---------------------------------------------------------------- layernorm + residual
__global__ __launch_bounds__(256) void ln_residual_kernel(
    const float* __restrict__ resid, const float* __restrict__ src,
    const float* __restrict__ gamma, const float* __restrict__ beta,
    float* __restrict__ outf, u16* __restrict__ outb) {
  __shared__ float red[2][256];
  const int t = threadIdx.x;
  const size_t base = (size_t)blockIdx.x * DM;

  float xv[4], lsum = 0.0f, lsq = 0.0f;
#pragma unroll
  for (int j = 0; j < 4; ++j) {
    const float s = src[base + t + j * 256];
    xv[j] = s; lsum += s; lsq += s * s;
  }
  red[0][t] = lsum; red[1][t] = lsq;
  __syncthreads();
  for (int off = 128; off > 0; off >>= 1) {
    if (t < off) { red[0][t] += red[0][t + off]; red[1][t] += red[1][t + off]; }
    __syncthreads();
  }
  const float mean = red[0][0] * (1.0f / DM);
  const float var  = red[1][0] * (1.0f / DM) - mean * mean;
  const float rstd = rsqrtf(var + EPSF);
#pragma unroll
  for (int j = 0; j < 4; ++j) {
    const int c = t + j * 256;
    const float y = (xv[j] - mean) * rstd * gamma[c] + beta[c] + resid[base + c];
    if (outf) outf[base + c] = y;
    if (outb) outb[base + c] = f2bf(y);
  }
}

// ---------------------------------------------------------------- launch
extern "C" void kernel_launch(void* const* d_in, const int* in_sizes, int n_in,
                              void* d_out, int out_size, void* d_ws, size_t ws_size,
                              hipStream_t stream) {
  (void)in_sizes; (void)n_in; (void)out_size; (void)ws_size;

  const float* x_enc = (const float*)d_in[0];
  const float* x_pos = (const float*)d_in[1];
  const float* wq = (const float*)d_in[2];
  const float* wk = (const float*)d_in[3];
  const float* wv = (const float*)d_in[4];
  const float* w1 = (const float*)d_in[5];
  const float* b1 = (const float*)d_in[6];
  const float* w2 = (const float*)d_in[7];
  const float* b2 = (const float*)d_in[8];
  const float* g1 = (const float*)d_in[9];
  const float* be1 = (const float*)d_in[10];
  const float* g2 = (const float*)d_in[11];
  const float* be2 = (const float*)d_in[12];
  float* out = (float*)d_out;

  char* p = (char*)d_ws;
  auto take = [&](size_t s) { char* r = p; p += (s + 255) & ~(size_t)255; return r; };
  const size_t szXF = (size_t)MROWS * DM * sizeof(float);
  const size_t szXB = (size_t)MROWS * DM * sizeof(u16);
  float* xf  = (float*)take(szXF);
  u16*   xb  = (u16*)  take(szXB);
  float* qb  = (float*)take(szXF);
  float* kb  = (float*)take(szXF);
  float* vb  = (float*)take(szXF);
  float* hf  = (float*)take(szXF);
  u16*   hb  = (u16*)  take(szXB);
  u16*   f1b = (u16*)  take((size_t)MROWS * DF * sizeof(u16));
  float* f2f = (float*)take(szXF);
  u16* wqt = (u16*)take((size_t)DM * DM * 2);   // all weights stored TRANSPOSED [N][K]
  u16* wkt = (u16*)take((size_t)DM * DM * 2);
  u16* wvt = (u16*)take((size_t)DM * DM * 2);
  u16* w1t = (u16*)take((size_t)DM * DF * 2);
  u16* w2t = (u16*)take((size_t)DF * DM * 2);

  const size_t nXD = (size_t)MROWS * DM;

  // 1) x = x_enc + x_pos (f32 + bf16); weights -> bf16 transposed [N][K]
  add_cvt_kernel<<<2048, 256, 0, stream>>>(x_enc, x_pos, xf, xb, nXD);
  cvt_transpose_kernel<<<dim3(DM / 64, DM / 64), 256, 0, stream>>>(wq, wqt, DM, DM);
  cvt_transpose_kernel<<<dim3(DM / 64, DM / 64), 256, 0, stream>>>(wk, wkt, DM, DM);
  cvt_transpose_kernel<<<dim3(DM / 64, DM / 64), 256, 0, stream>>>(wv, wvt, DM, DM);
  cvt_transpose_kernel<<<dim3(DF / 64, DM / 64), 256, 0, stream>>>(w1, w1t, DM, DF);
  cvt_transpose_kernel<<<dim3(DM / 64, DF / 64), 256, 0, stream>>>(w2, w2t, DF, DM);

  // 2) q/k/v projections (q,k fused elu+1)
  dim3 gD(DM / BKN, MROWS / BKM);       // (4, 128)
  gemm_bf16_kernel<<<gD, 256, 0, stream>>>(xb, wqt, nullptr, qb, nullptr, MROWS, DM, DM, 1);
  gemm_bf16_kernel<<<gD, 256, 0, stream>>>(xb, wkt, nullptr, kb, nullptr, MROWS, DM, DM, 1);
  gemm_bf16_kernel<<<gD, 256, 0, stream>>>(xb, wvt, nullptr, vb, nullptr, MROWS, DM, DM, 0);

  // 3) prefix-sum linear attention (attn written over q)
  scan_kernel<<<(NBATCH * DM) / 256, 256, 0, stream>>>(qb, kb, vb, qb);

  // 4) h = x + LN(attn)
  ln_residual_kernel<<<MROWS, 256, 0, stream>>>(xf, qb, g1, be1, hf, hb);

  // 5) FFN1: f1 = relu(h W1 + b1)^2  (bf16 out)
  dim3 gF(DF / BKN, MROWS / BKM);       // (16, 128)
  gemm_bf16_kernel<<<gF, 256, 0, stream>>>(hb, w1t, b1, nullptr, f1b, MROWS, DF, DM, 2);

  // 6) FFN2: f2 = f1 W2 + b2  (f32 out)
  gemm_bf16_kernel<<<gD, 256, 0, stream>>>(f1b, w2t, b2, f2f, nullptr, MROWS, DM, DF, 0);

  // 7) out = h + LN(f2)
  ln_residual_kernel<<<MROWS, 256, 0, stream>>>(hf, f2f, g2, be2, out, nullptr);
}
